// AdaptiveAttention_83322365542603
// MI455X (gfx1250) — compile-verified, matrix operand order checked by reading, not run on hardware
//
#include <hip/hip_runtime.h>
#include <hip/hip_bf16.h>
#include <math.h>

// ---------------------------------------------------------------------------
// AdaptiveAttention on gfx1250 (MI455X): bf16 WMMA pipeline, round 3
//   - block-cooperative LDS double-buffered weight staging in GEMMs
//   - async global->LDS DMA (ASYNCcnt) via global_load_async_to_lds_b128
//   - software-pipelined flash attention (K prefetch over softmax VALU)
// ---------------------------------------------------------------------------

#define D_MODEL 1024
#define N_HEADS 16
#define BATCH   2
#define SEQ     2048
#define DK      64

typedef __attribute__((ext_vector_type(16))) __bf16 v16bf;
typedef __attribute__((ext_vector_type(8)))  __bf16 v8bf;
typedef __attribute__((ext_vector_type(4)))  __bf16 v4bf;
typedef __attribute__((ext_vector_type(8)))  float  v8f;
typedef __attribute__((ext_vector_type(4)))  int    v4i;

// address-space-qualified pointee types for the async-DMA builtin
typedef __attribute__((address_space(1))) v4i v4i_g;   // global
typedef __attribute__((address_space(3))) v4i v4i_l;   // LDS

#if defined(__has_builtin)
#if __has_builtin(__builtin_amdgcn_global_load_async_to_lds_b128) && \
    __has_builtin(__builtin_amdgcn_s_wait_asynccnt)
#define AA_ASYNC 1
#endif
#endif
#ifndef AA_ASYNC
#define AA_ASYNC 0
#endif

__device__ __forceinline__ v8f vzero8() {
  v8f z = {0.f, 0.f, 0.f, 0.f, 0.f, 0.f, 0.f, 0.f};
  return z;
}

__device__ __forceinline__ v16bf ldfrag(const __bf16* p0, const __bf16* p1) {
  union { v16bf v; v8bf h[2]; } u;
  u.h[0] = *(const v8bf*)p0;
  u.h[1] = *(const v8bf*)p1;
  return u.v;
}

__device__ __forceinline__ v8f wmma_bf16(v16bf a, v16bf b, v8f c) {
  return __builtin_amdgcn_wmma_f32_16x16x32_bf16(
      /*neg_a=*/false, a, /*neg_b=*/false, b,
      /*c_mod=*/(short)0, c, /*reuse_a=*/false, /*reuse_b=*/false);
}

// Stage one 16-byte chunk global -> LDS. Async DMA path when available.
__device__ __forceinline__ void stage16(const __bf16* gsrc, __bf16* ldst) {
#if AA_ASYNC
  __builtin_amdgcn_global_load_async_to_lds_b128(
      (v4i_g*)gsrc, (v4i_l*)ldst, /*imm offset=*/0, /*cpol=*/0);
#else
  *(v8bf*)ldst = *(const v8bf*)gsrc;
#endif
}

__device__ __forceinline__ void wait_async_all() {
#if AA_ASYNC
  __builtin_amdgcn_s_wait_asynccnt(0);
#endif
}

// ---------------------------------------------------------------------------
// fp32 -> bf16 conversion (vectorized x4)
// ---------------------------------------------------------------------------
__global__ void aa_cvt_bf16x4(const float4* __restrict__ src,
                              v4bf* __restrict__ dst, int n4) {
  int i = blockIdx.x * blockDim.x + threadIdx.x;
  if (i < n4) {
    float4 f = src[i];
    v4bf o;
    o.x = (__bf16)f.x; o.y = (__bf16)f.y;
    o.z = (__bf16)f.z; o.w = (__bf16)f.w;
    dst[i] = o;
  }
}

// ---------------------------------------------------------------------------
// GEMM: Y[m,n] = sum_k X[m,k]*W[n,k] + bias[n]
//   Block: 8 waves -> 128x64 tile; W slab (64x32) double-buffered in LDS.
//   mode 0: dst bf16 [((b*H+h)*S+s)*64+dd]  (Q,K head-major)
//   mode 1: dst bf16 [((b*H+h)*64+dd)*S+s]  (V transposed)
//   mode 2: dst fp32 row-major [M, D_MODEL]
// ---------------------------------------------------------------------------
#define WROW 40  // padded LDS row stride in halves (32 data + 8 pad)

__global__ void aa_gemm(const __bf16* __restrict__ X,
                        const __bf16* __restrict__ W,
                        const float*  __restrict__ bias,
                        __bf16* __restrict__ dstb,
                        float*  __restrict__ dstf, int mode) {
  __shared__ alignas(16) __bf16 wsl[2][64 * WROW];

  const int lane  = threadIdx.x & 31;
  const int wid   = threadIdx.x >> 5;
  const int mtile = blockIdx.x >> 4;            // 0..31  (128 rows each)
  const int nbase = (blockIdx.x & 15) << 6;     // 0..960 (64 cols)
  const int mrow  = mtile * 128 + wid * 16;
  const int m     = lane & 15;
  const int g     = lane >> 4;

  // staging assignment: 256 threads cover 64 rows x 4 chunks of 8 halves
  const int srow   = threadIdx.x >> 2;          // 0..63
  const int schunk = (threadIdx.x & 3) * 8;     // half offset 0,8,16,24
  const __bf16* wsrc = W + (size_t)(nbase + srow) * D_MODEL + schunk;
  __bf16* wdst0 = &wsl[0][srow * WROW + schunk];
  __bf16* wdst1 = &wsl[1][srow * WROW + schunk];

  const __bf16* xrow = X + (size_t)(mrow + m) * D_MODEL;

  v8f c[4] = {vzero8(), vzero8(), vzero8(), vzero8()};

  stage16(wsrc, wdst0);  // prologue: k0 = 0 into buffer 0

  for (int i = 0; i < D_MODEL / 32; i++) {
    wait_async_all();
    __syncthreads();
    if (i + 1 < D_MODEL / 32)
      stage16(wsrc + (i + 1) * 32, ((i + 1) & 1) ? wdst1 : wdst0);

    const __bf16* buf = &wsl[i & 1][0];
    v16bf a = ldfrag(xrow + i * 32 + 8 * g, xrow + i * 32 + 16 + 8 * g);
#pragma unroll
    for (int j = 0; j < 4; j++) {
      const __bf16* wr = buf + (j * 16 + m) * WROW;
      c[j] = wmma_bf16(a, ldfrag(wr + 8 * g, wr + 16 + 8 * g), c[j]);
    }
  }

#pragma unroll
  for (int j = 0; j < 4; j++) {
    const int col = nbase + j * 16 + m;
    const float bv = bias[col];
    const int h  = col >> 6;
    const int dd = col & 63;
#pragma unroll
    for (int r = 0; r < 8; r++) {
      const int M = mrow + r + 8 * g;
      const float y = c[j][r] + bv;
      if (mode == 2) {
        dstf[(size_t)M * D_MODEL + col] = y;
      } else {
        const int b = M >> 11;          // / SEQ
        const int s = M & (SEQ - 1);
        size_t idx;
        if (mode == 1)
          idx = ((size_t)(b * N_HEADS + h) * DK + dd) * SEQ + s;
        else
          idx = ((size_t)(b * N_HEADS + h) * SEQ + s) * DK + dd;
        dstb[idx] = (__bf16)y;
      }
    }
  }
}

// ---------------------------------------------------------------------------
// Fused attention, flash-style online softmax, software pipelined.
//   scores = (Q K^T / 8) * scale * exp(-td_h * keypos)
//   attn   = softmax(scores) * sigmoid(scores)
//   ctx    = attn @ V  -> bf16 [B,S,D_MODEL]
// One wave owns 16 queries of one (b,h); loops keys in blocks of 32.
// ---------------------------------------------------------------------------
#define TROW 40  // padded transpose-buffer row stride in halves

__global__ void aa_attn(const __bf16* __restrict__ Qh,
                        const __bf16* __restrict__ Kh,
                        const __bf16* __restrict__ Vt,
                        const float* __restrict__ scale_p,
                        const float* __restrict__ td_p,
                        __bf16* __restrict__ ctx) {
  __shared__ alignas(16) __bf16 wbuf[8][16 * TROW];

  const int lane = threadIdx.x & 31;
  const int wid  = threadIdx.x >> 5;
  const int t    = blockIdx.x * 8 + wid;          // 0..4095
  const int qt   = t & 127;
  const int h    = (t >> 7) & (N_HEADS - 1);
  const int b    = t >> 11;
  const int qbase = qt << 4;
  const int m = lane & 15;
  const int g = lane >> 4;

  const float sc = 0.125f * scale_p[0];
  const float td = td_p[h];

  const __bf16* Qb = Qh + (size_t)(b * N_HEADS + h) * SEQ * DK;
  const __bf16* Kb = Kh + (size_t)(b * N_HEADS + h) * SEQ * DK;
  const __bf16* Vb = Vt + (size_t)(b * N_HEADS + h) * DK * SEQ;

  const __bf16* qr = Qb + (size_t)(qbase + m) * DK;
  const v16bf qf0 = ldfrag(qr + 8 * g,      qr + 16 + 8 * g);
  const v16bf qf1 = ldfrag(qr + 32 + 8 * g, qr + 48 + 8 * g);

  v8f acc[4] = {vzero8(), vzero8(), vzero8(), vzero8()};
  float mrun[8], lrun[8];
#pragma unroll
  for (int r = 0; r < 8; r++) { mrun[r] = -INFINITY; lrun[r] = 0.f; }

  // prime K fragments for kb = 0
  v16bf kc[4];
  {
    const __bf16* k0 = Kb + (size_t)m * DK;
    const __bf16* k1 = Kb + (size_t)(16 + m) * DK;
    kc[0] = ldfrag(k0 + 8 * g,      k0 + 16 + 8 * g);
    kc[1] = ldfrag(k0 + 32 + 8 * g, k0 + 48 + 8 * g);
    kc[2] = ldfrag(k1 + 8 * g,      k1 + 16 + 8 * g);
    kc[3] = ldfrag(k1 + 32 + 8 * g, k1 + 48 + 8 * g);
  }

  for (int kb = 0; kb < SEQ; kb += 32) {
    // ---- scores (uses resident K fragments) ----
    v8f s0 = vzero8(), s1 = vzero8();
    s0 = wmma_bf16(qf0, kc[0], s0);
    s0 = wmma_bf16(qf1, kc[1], s0);
    s1 = wmma_bf16(qf0, kc[2], s1);
    s1 = wmma_bf16(qf1, kc[3], s1);

    // ---- prefetch next block's K and this block's V; latency hidden by
    //      the softmax VALU section below ----
    const int kbn = (kb + 32 < SEQ) ? kb + 32 : 0;
    v16bf kn[4];
    {
      const __bf16* k0 = Kb + (size_t)(kbn + m) * DK;
      const __bf16* k1 = Kb + (size_t)(kbn + 16 + m) * DK;
      kn[0] = ldfrag(k0 + 8 * g,      k0 + 16 + 8 * g);
      kn[1] = ldfrag(k0 + 32 + 8 * g, k0 + 48 + 8 * g);
      kn[2] = ldfrag(k1 + 8 * g,      k1 + 16 + 8 * g);
      kn[3] = ldfrag(k1 + 32 + 8 * g, k1 + 48 + 8 * g);
    }
    v16bf vf[4];
#pragma unroll
    for (int j = 0; j < 4; j++) {
      const __bf16* vr = Vb + (size_t)(j * 16 + m) * SEQ + kb;
      vf[j] = ldfrag(vr + 8 * g, vr + 16 + 8 * g);
    }

    // ---- scale, decay, online softmax * sigmoid ----
    const float d0 = __expf(-td * (float)(kb + m));
    const float d1 = __expf(-td * (float)(kb + 16 + m));
#pragma unroll
    for (int r = 0; r < 8; r++) {
      const float a0 = s0[r] * sc * d0;
      const float a1 = s1[r] * sc * d1;
      float mx = fmaxf(a0, a1);
      mx = fmaxf(mx, __shfl_xor(mx, 1));
      mx = fmaxf(mx, __shfl_xor(mx, 2));
      mx = fmaxf(mx, __shfl_xor(mx, 4));
      mx = fmaxf(mx, __shfl_xor(mx, 8));
      const float mnew = fmaxf(mrun[r], mx);
      const float resc = __expf(mrun[r] - mnew);
      const float p0 = __expf(a0 - mnew);
      const float p1 = __expf(a1 - mnew);
      const float w0 = p0 * (1.f / (1.f + __expf(-a0)));
      const float w1 = p1 * (1.f / (1.f + __expf(-a1)));
      float ps = p0 + p1;
      ps += __shfl_xor(ps, 1);
      ps += __shfl_xor(ps, 2);
      ps += __shfl_xor(ps, 4);
      ps += __shfl_xor(ps, 8);
      lrun[r] = lrun[r] * resc + ps;
      mrun[r] = mnew;
      acc[0][r] *= resc; acc[1][r] *= resc;
      acc[2][r] *= resc; acc[3][r] *= resc;
      wbuf[wid][(r + 8 * g) * TROW + m]      = (__bf16)w0;
      wbuf[wid][(r + 8 * g) * TROW + 16 + m] = (__bf16)w1;
    }
    __builtin_amdgcn_wave_barrier();

    // weights C-layout -> A-layout via LDS transpose
    v16bf wa = ldfrag(&wbuf[wid][m * TROW + 8 * g],
                      &wbuf[wid][m * TROW + 16 + 8 * g]);
    __builtin_amdgcn_wave_barrier();

#pragma unroll
    for (int j = 0; j < 4; j++)
      acc[j] = wmma_bf16(wa, vf[j], acc[j]);

#pragma unroll
    for (int j = 0; j < 4; j++) kc[j] = kn[j];
  }

  // epilogue: divide by softmax denominator, store ctx bf16 [B,S,D]
#pragma unroll
  for (int j = 0; j < 4; j++) {
#pragma unroll
    for (int r = 0; r < 8; r++) {
      const int s   = qbase + r + 8 * g;
      const int col = h * DK + j * 16 + m;
      const float v = acc[j][r] / lrun[r];
      ctx[(size_t)(b * SEQ + s) * D_MODEL + col] = (__bf16)v;
    }
  }
}

// ---------------------------------------------------------------------------
// Host launch
// ---------------------------------------------------------------------------
extern "C" void kernel_launch(void* const* d_in, const int* in_sizes, int n_in,
                              void* d_out, int out_size, void* d_ws, size_t ws_size,
                              hipStream_t stream) {
  (void)in_sizes; (void)n_in; (void)out_size; (void)ws_size;

  const float* q  = (const float*)d_in[0];
  const float* k  = (const float*)d_in[1];
  const float* v  = (const float*)d_in[2];
  const float* wq = (const float*)d_in[3];
  const float* bq = (const float*)d_in[4];
  const float* wk = (const float*)d_in[5];
  const float* bk = (const float*)d_in[6];
  const float* wv = (const float*)d_in[7];
  const float* bv = (const float*)d_in[8];
  const float* wo = (const float*)d_in[9];
  const float* bo = (const float*)d_in[10];
  const float* scale = (const float*)d_in[11];
  const float* td    = (const float*)d_in[12];
  float* out = (float*)d_out;

  const size_t NELM = (size_t)BATCH * SEQ * D_MODEL;   // 4,194,304
  const size_t WELM = (size_t)D_MODEL * D_MODEL;       // 1,048,576

  char* ws = (char*)d_ws;
  size_t off = 0;
  __bf16* qb  = (__bf16*)(ws + off); off += NELM * 2;
  __bf16* kb  = (__bf16*)(ws + off); off += NELM * 2;
  __bf16* vb  = (__bf16*)(ws + off); off += NELM * 2;
  __bf16* wqb = (__bf16*)(ws + off); off += WELM * 2;
  __bf16* wkb = (__bf16*)(ws + off); off += WELM * 2;
  __bf16* wvb = (__bf16*)(ws + off); off += WELM * 2;
  __bf16* wob = (__bf16*)(ws + off); off += WELM * 2;
  __bf16* Qh  = (__bf16*)(ws + off); off += NELM * 2;
  __bf16* Kh  = (__bf16*)(ws + off); off += NELM * 2;
  __bf16* Vt  = (__bf16*)(ws + off); off += NELM * 2;
  __bf16* ctx = (__bf16*)(ws + off); off += NELM * 2;

  // 1) convert to bf16
  {
    int n4 = (int)(NELM / 4);
    int blocks = (n4 + 255) / 256;
    aa_cvt_bf16x4<<<blocks, 256, 0, stream>>>((const float4*)q, (v4bf*)qb, n4);
    aa_cvt_bf16x4<<<blocks, 256, 0, stream>>>((const float4*)k, (v4bf*)kb, n4);
    aa_cvt_bf16x4<<<blocks, 256, 0, stream>>>((const float4*)v, (v4bf*)vb, n4);
    int w4 = (int)(WELM / 4);
    int wblocks = (w4 + 255) / 256;
    aa_cvt_bf16x4<<<wblocks, 256, 0, stream>>>((const float4*)wq, (v4bf*)wqb, w4);
    aa_cvt_bf16x4<<<wblocks, 256, 0, stream>>>((const float4*)wk, (v4bf*)wkb, w4);
    aa_cvt_bf16x4<<<wblocks, 256, 0, stream>>>((const float4*)wv, (v4bf*)wvb, w4);
    aa_cvt_bf16x4<<<wblocks, 256, 0, stream>>>((const float4*)wo, (v4bf*)wob, w4);
  }

  // 2) QKV projections: 512 blocks x 256 threads, 128x64 tile per block
  aa_gemm<<<512, 256, 0, stream>>>(qb, wqb, bq, Qh, nullptr, 0);
  aa_gemm<<<512, 256, 0, stream>>>(kb, wkb, bk, Kh, nullptr, 0);
  aa_gemm<<<512, 256, 0, stream>>>(vb, wvb, bv, Vt, nullptr, 1);

  // 3) fused attention
  aa_attn<<<512, 256, 0, stream>>>(Qh, Kh, Vt, scale, td, ctx);

  // 4) output projection -> fp32
  aa_gemm<<<512, 256, 0, stream>>>(ctx, wob, bo, nullptr, out, 2);
}
